// LOPModel_67577015435429
// MI455X (gfx1250) — matrix operand order repaired and checked by reading, not compiled, not run on hardware
//
#include <hip/hip_runtime.h>
#include <hip/hip_bf16.h>

// ---------------------------------------------------------------------------
// LOP pointer-network decoder for MI455X (gfx1250, wave32, WMMA).
//  * e_layers (3 x 41MB) precomputed once with WMMA GEMMs; L2-resident (192MB).
//  * GEMMs: v_wmma_f32_16x16x32_f16, 4 M-tiles (64 rows) per wave sharing one
//    pre-swizzled fragment-major B fragment (2 x b128 loads per K-chunk).
//  * Irregular ops fused into few VALU kernels; sequential 49-step decode.
// ---------------------------------------------------------------------------

typedef __attribute__((ext_vector_type(16))) _Float16 v16h;
typedef __attribute__((ext_vector_type(8)))  _Float16 h8;
typedef __attribute__((ext_vector_type(8)))  float    v8f;
typedef __attribute__((ext_vector_type(4)))  float    f4;

constexpr int kB = 32, kN = 50, kD = 128, kH = 8, kHD = 16, kFF = 512, kT = kN - 1;

// ---------------- weight swizzle: W_f32[K][Nc] -> fragment-major f16 ---------
// out[((ntile*kc + chunk)*32 + lane)*16 + i] = W[chunk*32 + (lane>>4)*16 + i][ntile*16 + (lane&15)]
// Optionally fuses 3 side-by-side weight matrices (each K x seg, Nc = 3*seg).
__global__ void k_build_wswz3(const float* __restrict__ W0, const float* __restrict__ W1,
                              const float* __restrict__ W2, _Float16* __restrict__ out,
                              int K, int Nc, int seg) {
  int idx = blockIdx.x * blockDim.x + threadIdx.x;
  if (idx >= K * Nc) return;
  int i     = idx & 15;
  int lane  = (idx >> 4) & 31;
  int rest  = idx >> 9;                   // chunk + ntile*kc
  int kc    = K >> 5;
  int chunk = rest % kc;
  int ntile = rest / kc;
  int krow  = chunk * 32 + ((lane >> 4) << 4) + i;
  int ncol  = ntile * 16 + (lane & 15);
  const float* src = W0; int c = ncol;
  if (W1 && ncol >= seg) {
    if (ncol < 2 * seg) { src = W1; c = ncol - seg; }
    else                { src = W2; c = ncol - 2 * seg; }
  }
  out[idx] = (_Float16)src[(size_t)krow * seg + c];
}

__global__ void k_concat3(const float* __restrict__ a, const float* __restrict__ b,
                          const float* __restrict__ c, float* __restrict__ o, int n) {
  int i = blockIdx.x * blockDim.x + threadIdx.x;
  if (i < n) o[i] = a[i];
  else if (i < 2 * n) o[i] = b[i - n];
  else if (i < 3 * n) o[i] = c[i - 2 * n];
}

__global__ void k_edge_embed(const float* __restrict__ edges, const float* __restrict__ ew,
                             const float* __restrict__ eb, float* __restrict__ out) {
  long i = (long)blockIdx.x * blockDim.x + threadIdx.x;
  const long tot = (long)kB * kN * kN * kD;
  if (i >= tot) return;
  int d = (int)(i % kD);
  long bij = i / kD;
  out[i] = edges[bij] * ew[d] + eb[d];
}

__global__ void k_init_state(float* mask, float* sel) {
  int i = blockIdx.x * blockDim.x + threadIdx.x;
  if (i < kB * kN) { mask[i] = 0.f; sel[i] = 0.f; }
}

__global__ void k_encode_init(const float* __restrict__ sel, const float* __restrict__ node_w,
                              const float* __restrict__ node_b, float* __restrict__ h) {
  int i = blockIdx.x * blockDim.x + threadIdx.x;
  if (i >= kB * kN * kD) return;
  int d = i % kD;
  int bn = i / kD;
  float s = sel[bn];
  h[i] = (1.0f - s) * node_w[d] + s * node_w[kD + d] + node_b[d];
}

// ---------------- WMMA GEMM ------------------------------------------------
// C[M,Nc] = act(A_f32[M,K] @ W + bias), W pre-swizzled fragment-major f16.
// One wave32 computes a 64x16 C block: 4 M-tiles sharing one B fragment.
// M % 64 == 0, K % 32 == 0, Nc % 16 == 0.
__global__ void __launch_bounds__(256)
k_gemm_wmma(const float* __restrict__ A, const _Float16* __restrict__ Wsw,
            const float* __restrict__ bias, float* __restrict__ C,
            int M, int K, int Nc, int act) {
  int gwave = (int)(((long)blockIdx.x * blockDim.x + threadIdx.x) >> 5);
  int lane  = threadIdx.x & 31;
  int nt    = Nc >> 4;
  int total = (M >> 6) * nt;
  if (gwave >= total) return;             // per-wave uniform: EXEC all-1 for WMMA
  int tm  = (gwave / nt) << 6;            // 64-row superblock base
  int tn  = gwave % nt;                   // 16-col tile index
  int l16 = lane & 15;
  int hi  = lane >> 4;
  int aoff = hi << 4;                     // A-fragment K offset 0/16
  int kc   = K >> 5;
  v8f acc[4] = {{}, {}, {}, {}};
  const _Float16* wbase = Wsw + ((size_t)tn * kc * 32 + lane) * 16;
  const float*    a0    = A + (size_t)(tm + l16) * K + aoff;
  const size_t rstride  = (size_t)16 * K; // one M-tile down

  for (int c = 0; c < kc; ++c) {
    const _Float16* wp = wbase + (size_t)c * 512;      // 32 lanes * 16 halves
    h8 blo = *(const h8*)(wp);
    h8 bhi = *(const h8*)(wp + 8);
    v16h bf = __builtin_shufflevector(blo, bhi, 0, 1, 2, 3, 4, 5, 6, 7,
                                      8, 9, 10, 11, 12, 13, 14, 15);
    __builtin_prefetch((const void*)(wp + 512), 0, 1); // next K-chunk of W
    int k0 = c << 5;
#pragma unroll
    for (int m = 0; m < 4; ++m) {
      const float* ap = a0 + (size_t)m * rstride + k0;
      f4 x0 = *(const f4*)(ap);
      f4 x1 = *(const f4*)(ap + 4);
      f4 x2 = *(const f4*)(ap + 8);
      f4 x3 = *(const f4*)(ap + 12);
      __builtin_prefetch((const void*)(ap + 32), 0, 1);
      v16h af;
#pragma unroll
      for (int i = 0; i < 4; ++i) {
        af[i]      = (_Float16)x0[i];
        af[i + 4]  = (_Float16)x1[i];
        af[i + 8]  = (_Float16)x2[i];
        af[i + 12] = (_Float16)x3[i];
      }
      acc[m] = __builtin_amdgcn_wmma_f32_16x16x32_f16(false, af, false, bf,
                                                      (short)0, acc[m], false, false);
    }
  }
  int ncol = tn * 16 + l16;
  float bv = bias ? bias[ncol] : 0.f;
#pragma unroll
  for (int m = 0; m < 4; ++m) {
    int rbase = tm + m * 16 + (hi << 3);
#pragma unroll
    for (int r = 0; r < 8; ++r) {
      float v = acc[m][r] + bv;
      if (act == 1) {                     // new-GELU
        float xx = v, x3 = xx * xx * xx;
        v = 0.5f * xx * (1.f + tanhf(0.7978845608028654f * (xx + 0.044715f * x3)));
      }
      C[(size_t)(rbase + r) * Nc + ncol] = v;
    }
  }
}

// ---------------- fused edge attention: scores + softmax(heads) + A@V --------
// block per (b,i), 128 threads. qkv: [B*N][384] (q|k|v), e: [B,N,N,128].
__global__ void k_attn_fused(const float* __restrict__ qkv, const float* __restrict__ e,
                             const float* __restrict__ attn_w, const float* __restrict__ attn_b,
                             float* __restrict__ x) {
  __shared__ float aw[kHD];
  __shared__ float sc[kH][kN];
  int b = blockIdx.x / kN, i = blockIdx.x % kN;
  float ab = attn_b[0];
  if (threadIdx.x < kHD) aw[threadIdx.x] = attn_w[threadIdx.x];
  __syncthreads();
  const float* qrow = qkv + (size_t)(b * kN + i) * (3 * kD);
  for (int idx = threadIdx.x; idx < kH * kN; idx += blockDim.x) {
    int h = idx / kN, j = idx % kN;
    const float* qp = qrow + h * kHD;
    const float* kp = qkv + (size_t)(b * kN + j) * (3 * kD) + kD + h * kHD;
    const float* ep = e + ((size_t)(b * kN + i) * kN + j) * kD + h * kHD;
    float s = 0.f;
#pragma unroll
    for (int d = 0; d < kHD; ++d) s += qp[d] * kp[d] * ep[d] * aw[d];
    sc[h][j] = s * 0.25f + ab;            // / sqrt(HEAD_DIM)
  }
  __syncthreads();
  if (threadIdx.x < kN) {                 // softmax over heads (axis=1 of bhij)
    int j = threadIdx.x;
    float mx = -__builtin_inff();
    for (int h = 0; h < kH; ++h) mx = fmaxf(mx, sc[h][j]);
    float sum = 0.f;
    for (int h = 0; h < kH; ++h) { float ev = expf(sc[h][j] - mx); sc[h][j] = ev; sum += ev; }
    float r = 1.f / sum;
    for (int h = 0; h < kH; ++h) sc[h][j] *= r;
  }
  __syncthreads();
  int h = threadIdx.x >> 4;               // 128 threads = (h,hd)
  const float* vp = qkv + (size_t)b * kN * (3 * kD) + 2 * kD + threadIdx.x;
  float acc = 0.f;
  for (int j = 0; j < kN; ++j) acc += sc[h][j] * vp[(size_t)j * (3 * kD)];
  x[(size_t)(b * kN + i) * kD + threadIdx.x] = acc;
}

// ---------------- residual add + instance-norm over tokens (axis=1) ----------
__global__ void k_add_inorm(const float* __restrict__ base, const float* __restrict__ delta,
                            const float* __restrict__ g, const float* __restrict__ bb,
                            float* __restrict__ out) {
  int idx = blockIdx.x * blockDim.x + threadIdx.x;   // B*D
  if (idx >= kB * kD) return;
  int b = idx / kD, d = idx % kD;
  const float* p1 = base + (size_t)b * kN * kD + d;
  const float* p2 = delta + (size_t)b * kN * kD + d;
  float s = 0.f, ss = 0.f;
  for (int n = 0; n < kN; ++n) { float y = p1[(size_t)n * kD] + p2[(size_t)n * kD]; s += y; ss += y * y; }
  float m = s / (float)kN;
  float var = ss / (float)kN - m * m;
  float inv = rsqrtf(var + 1e-5f);
  float gg = g[d], bv = bb[d];
  for (int n = 0; n < kN; ++n) {
    float y = p1[(size_t)n * kD] + p2[(size_t)n * kD];
    out[((size_t)b * kN + n) * kD + d] = (y - m) * inv * gg + bv;
  }
}

// ---------------- fused pointer/decode step (block per batch row, 128 thr) ---
__global__ void k_decode(const float* __restrict__ h, const float* __restrict__ kv,
                         const float* __restrict__ placeholder,
                         const float* __restrict__ wq2_w, const float* __restrict__ wq2_b,
                         const float* __restrict__ comb_w, const float* __restrict__ comb_b,
                         float* mask, float* sel, float* prob_out, int* sol_out, int t) {
  __shared__ float ctx[2 * kD];
  __shared__ float q2[kD];
  __shared__ float sw[kH][kN];
  __shared__ float outv[kD];
  __shared__ float mh[kD];
  __shared__ float logits[kN];
  __shared__ float selbuf[kN];
  int b = blockIdx.x;
  int tid = threadIdx.x;                  // 128
  if (tid < kN) selbuf[tid] = sel[b * kN + tid];
  __syncthreads();
  if (t == 0) {
    for (int c = tid; c < 2 * kD; c += blockDim.x) ctx[c] = placeholder[c];
  } else {
    const float* hb = h + (size_t)b * kN * kD + tid;
    float s = 0.f, ssel = 0.f;
    for (int n = 0; n < kN; ++n) { float v = hb[(size_t)n * kD]; s += v; ssel += selbuf[n] * v; }
    ctx[tid]      = s / (float)kN;
    ctx[kD + tid] = ssel / (float)t;      // == / max(t,1) for t>=1
  }
  __syncthreads();
  {                                       // q2 = ctx @ wq2_w + wq2_b
    float s = wq2_b[tid];
    for (int c = 0; c < 2 * kD; ++c) s += ctx[c] * wq2_w[(size_t)c * kD + tid];
    q2[tid] = s;
  }
  __syncthreads();
  for (int idx = tid; idx < kH * kN; idx += blockDim.x) {   // scores
    int hh = idx / kN, n = idx % kN;
    const float* kp = kv + (size_t)(b * kN + n) * (2 * kD) + hh * kHD;
    const float* qp = q2 + hh * kHD;
    float s = 0.f;
#pragma unroll
    for (int d = 0; d < kHD; ++d) s += qp[d] * kp[d];
    sw[hh][n] = s * 0.25f;
  }
  __syncthreads();
  if (tid < kH) {                         // softmax over n
    float mx = -__builtin_inff();
    for (int n = 0; n < kN; ++n) mx = fmaxf(mx, sw[tid][n]);
    float sum = 0.f;
    for (int n = 0; n < kN; ++n) { float e = expf(sw[tid][n] - mx); sw[tid][n] = e; sum += e; }
    float r = 1.f / sum;
    for (int n = 0; n < kN; ++n) sw[tid][n] *= r;
  }
  __syncthreads();
  {                                       // weighted V
    int hh = tid >> 4;
    const float* vp = kv + (size_t)b * kN * (2 * kD) + kD + tid;
    float s = 0.f;
    for (int n = 0; n < kN; ++n) s += sw[hh][n] * vp[(size_t)n * (2 * kD)];
    outv[tid] = s;
  }
  __syncthreads();
  {                                       // mh = outv @ comb_w + comb_b
    float s = comb_b[tid];
    for (int c = 0; c < kD; ++c) s += outv[c] * comb_w[(size_t)c * kD + tid];
    mh[tid] = s;
  }
  __syncthreads();
  if (tid < kN) {                         // logits = CLIP*tanh(mh.h/sqrt(D)) + mask
    const float* hn = h + (size_t)(b * kN + tid) * kD;
    float s = 0.f;
    for (int d = 0; d < kD; ++d) s += mh[d] * hn[d];
    logits[tid] = 10.f * tanhf(s * 0.08838834764831845f) + mask[b * kN + tid];
  }
  __syncthreads();
  if (tid == 0) {                         // softmax prob + deterministic Gumbel-max
    float mx = -__builtin_inff();
    for (int n = 0; n < kN; ++n) mx = fmaxf(mx, logits[n]);
    float sum = 0.f;
    for (int n = 0; n < kN; ++n) sum += expf(logits[n] - mx);
    float best = -__builtin_inff(); int bi = 0;
    for (int n = 0; n < kN; ++n) {
      unsigned xh = (unsigned)t * 0x9E3779B9u ^ (unsigned)b * 0x85EBCA6Bu ^ (unsigned)n * 0xC2B2AE35u;
      xh ^= xh >> 16; xh *= 0x7feb352du; xh ^= xh >> 15; xh *= 0x846ca68bu; xh ^= xh >> 16;
      float u = ((float)(xh >> 8) + 0.5f) * (1.0f / 16777216.0f);
      float g = -logf(-logf(u));
      float s2 = logits[n] + g;           // masked entries are -inf -> never picked
      if (s2 > best) { best = s2; bi = n; }
    }
    prob_out[b * kN + t] = expf(logits[bi] - mx) / sum;
    sol_out[b * kN + t]  = bi;
    mask[b * kN + bi] = -__builtin_inff();
    sel[b * kN + bi]  = 1.0f;
  }
}

__global__ void k_finalize(const float* __restrict__ mask, float* prob_out, int* sol_out) {
  int b = blockIdx.x * blockDim.x + threadIdx.x;
  if (b >= kB) return;
  prob_out[b * kN + kN - 1] = 1.0f;
  float best = mask[b * kN]; int bi = 0;
  for (int n = 1; n < kN; ++n) { float v = mask[b * kN + n]; if (v > best) { best = v; bi = n; } }
  sol_out[b * kN + kN - 1] = bi;
}

// ---------------------------------------------------------------------------

static inline void gemm(const float* A, const _Float16* W, const float* bias, float* C,
                        int M, int K, int Nc, int act, hipStream_t s) {
  int waves = (M >> 6) * (Nc >> 4);       // 64x16 C block per wave
  int blocks = (waves + 7) / 8;           // 8 waves (256 thr) per block
  k_gemm_wmma<<<blocks, 256, 0, s>>>(A, W, bias, C, M, K, Nc, act);
}

extern "C" void kernel_launch(void* const* d_in, const int* in_sizes, int n_in,
                              void* d_out, int out_size, void* d_ws, size_t ws_size,
                              hipStream_t stream) {
  (void)out_size; (void)ws_size;
  const float* P[66] = {nullptr};
  for (int i = 0; i < n_in && i < 66; ++i) P[i] = (const float*)d_in[i];

  // Detect pytree flatten order: insertion-order vs jax sorted-key order.
  // insertion: [1]=node_w(256),[2]=node_b(128); sorted: [1]=comb_b(128),[2]=comb_w(16384)
  bool sorted = (n_in > 2 && in_sizes[2] == 16384);

  const float *edges = P[0];
  const float *node_w, *node_b, *edge_w, *edge_b, *placeholder;
  const float *wq2_w, *wq2_b, *wkv_w, *wkv_b, *comb_w, *comb_b;
  const float *Lwq[3], *Lwqb[3], *Lwk[3], *Lwkb[3], *Lwv[3], *Lwvb[3];
  const float *Lwe[3], *Lweb[3], *Law[3], *Lab[3];
  const float *Lm1[3], *Lm1b[3], *Lm2[3], *Lm2b[3];
  const float *Ln1g[3], *Ln1b[3], *Ln2g[3], *Ln2b[3];

  if (!sorted) {
    node_w = P[1]; node_b = P[2]; edge_w = P[3]; edge_b = P[4]; placeholder = P[5];
    wq2_w = P[6]; wq2_b = P[7]; wkv_w = P[8]; wkv_b = P[9]; comb_w = P[10]; comb_b = P[11];
    for (int l = 0; l < 3; ++l) {
      int o = 12 + l * 18;
      Lwq[l] = P[o + 0]; Lwqb[l] = P[o + 1]; Lwk[l] = P[o + 2]; Lwkb[l] = P[o + 3];
      Lwv[l] = P[o + 4]; Lwvb[l] = P[o + 5]; Lwe[l] = P[o + 6]; Lweb[l] = P[o + 7];
      Law[l] = P[o + 8]; Lab[l] = P[o + 9];
      Lm1[l] = P[o + 10]; Lm1b[l] = P[o + 11]; Lm2[l] = P[o + 12]; Lm2b[l] = P[o + 13];
      Ln1g[l] = P[o + 14]; Ln1b[l] = P[o + 15]; Ln2g[l] = P[o + 16]; Ln2b[l] = P[o + 17];
    }
  } else {
    comb_b = P[1]; comb_w = P[2]; edge_b = P[3]; edge_w = P[4];
    for (int l = 0; l < 3; ++l) {
      int o = 5 + l * 18;   // sorted: attn_b,attn_w,m1_b,m1_w,m2_b,m2_w,n1_b,n1_g,n2_b,n2_g,we_b,we_w,wk_b,wk_w,wq_b,wq_w,wv_b,wv_w
      Lab[l] = P[o + 0];  Law[l] = P[o + 1];
      Lm1b[l] = P[o + 2]; Lm1[l] = P[o + 3]; Lm2b[l] = P[o + 4]; Lm2[l] = P[o + 5];
      Ln1b[l] = P[o + 6]; Ln1g[l] = P[o + 7]; Ln2b[l] = P[o + 8]; Ln2g[l] = P[o + 9];
      Lweb[l] = P[o + 10]; Lwe[l] = P[o + 11]; Lwkb[l] = P[o + 12]; Lwk[l] = P[o + 13];
      Lwqb[l] = P[o + 14]; Lwq[l] = P[o + 15]; Lwvb[l] = P[o + 16]; Lwv[l] = P[o + 17];
    }
    node_b = P[59]; node_w = P[60]; placeholder = P[61];
    wkv_b = P[62]; wkv_w = P[63]; wq2_b = P[64]; wq2_w = P[65];
  }

  // ---- workspace bump allocator ----
  char* ws = (char*)d_ws;
  size_t off = 0;
  auto alloc = [&](size_t bytes) -> char* {
    char* p = ws + off; off += (bytes + 255) & ~(size_t)255; return p;
  };
  const size_t eElems = (size_t)kB * kN * kN * kD;
  float* e_layer[3];
  for (int l = 0; l < 3; ++l) e_layer[l] = (float*)alloc(eElems * 4);
  float* e_emb = (float*)alloc(eElems * 4);
  float* h    = (float*)alloc((size_t)kB * kN * kD * 4);
  float* qkv  = (float*)alloc((size_t)kB * kN * 3 * kD * 4);
  float* x    = (float*)alloc((size_t)kB * kN * kD * 4);
  float* out1 = (float*)alloc((size_t)kB * kN * kD * 4);
  float* t1   = (float*)alloc((size_t)kB * kN * kFF * 4);
  float* t2   = (float*)alloc((size_t)kB * kN * kD * 4);
  float* kv   = (float*)alloc((size_t)kB * kN * 2 * kD * 4);
  float* mask = (float*)alloc((size_t)kB * kN * 4);
  float* sel  = (float*)alloc((size_t)kB * kN * 4);
  _Float16* wqkv16[3]; float* bqkv[3]; _Float16* we16[3]; _Float16* m116[3]; _Float16* m216[3];
  for (int l = 0; l < 3; ++l) {
    wqkv16[l] = (_Float16*)alloc((size_t)kD * 3 * kD * 2);
    bqkv[l]   = (float*)alloc((size_t)3 * kD * 4);
    we16[l]   = (_Float16*)alloc((size_t)kD * kD * 2);
    m116[l]   = (_Float16*)alloc((size_t)kD * kFF * 2);
    m216[l]   = (_Float16*)alloc((size_t)kFF * kD * 2);
  }
  _Float16* wkv16 = (_Float16*)alloc((size_t)kD * 2 * kD * 2);

  float* prob_out = (float*)d_out;
  int*   sol_out  = (int*)((float*)d_out + kB * kN);

  // ---- one-time setup (per launch; deterministic) ----
  for (int l = 0; l < 3; ++l) {
    k_build_wswz3<<<(kD * 3 * kD + 255) / 256, 256, 0, stream>>>(Lwq[l], Lwk[l], Lwv[l],
                                                                 wqkv16[l], kD, 3 * kD, kD);
    k_concat3<<<(3 * kD + 255) / 256, 256, 0, stream>>>(Lwqb[l], Lwkb[l], Lwvb[l], bqkv[l], kD);
    k_build_wswz3<<<(kD * kD + 255) / 256, 256, 0, stream>>>(Lwe[l], nullptr, nullptr,
                                                             we16[l], kD, kD, kD);
    k_build_wswz3<<<(kD * kFF + 255) / 256, 256, 0, stream>>>(Lm1[l], nullptr, nullptr,
                                                              m116[l], kD, kFF, kFF);
    k_build_wswz3<<<(kFF * kD + 255) / 256, 256, 0, stream>>>(Lm2[l], nullptr, nullptr,
                                                              m216[l], kFF, kD, kD);
  }
  k_build_wswz3<<<(kD * 2 * kD + 255) / 256, 256, 0, stream>>>(wkv_w, nullptr, nullptr,
                                                               wkv16, kD, 2 * kD, 2 * kD);

  k_edge_embed<<<(int)((eElems + 255) / 256), 256, 0, stream>>>(edges, edge_w, edge_b, e_emb);
  for (int l = 0; l < 3; ++l)
    gemm(e_emb, we16[l], Lweb[l], e_layer[l], kB * kN * kN, kD, kD, 0, stream); // 123MB, L2-resident

  k_init_state<<<(kB * kN + 255) / 256, 256, 0, stream>>>(mask, sel);

  // ---- 49 sequential decode steps ----
  for (int t = 0; t < kT; ++t) {
    k_encode_init<<<(kB * kN * kD + 255) / 256, 256, 0, stream>>>(sel, node_w, node_b, h);
    for (int l = 0; l < 3; ++l) {
      gemm(h, wqkv16[l], bqkv[l], qkv, kB * kN, kD, 3 * kD, 0, stream);
      k_attn_fused<<<kB * kN, 128, 0, stream>>>(qkv, e_layer[l], Law[l], Lab[l], x);
      k_add_inorm<<<(kB * kD + 255) / 256, 256, 0, stream>>>(h, x, Ln1g[l], Ln1b[l], out1);
      gemm(out1, m116[l], Lm1b[l], t1, kB * kN, kD, kFF, 1, stream);   // fused GELU
      gemm(t1, m216[l], Lm2b[l], t2, kB * kN, kFF, kD, 0, stream);
      k_add_inorm<<<(kB * kD + 255) / 256, 256, 0, stream>>>(out1, t2, Ln2g[l], Ln2b[l], h);
    }
    gemm(h, wkv16, wkv_b, kv, kB * kN, kD, 2 * kD, 0, stream);
    k_decode<<<kB, 128, 0, stream>>>(h, kv, placeholder, wq2_w, wq2_b, comb_w, comb_b,
                                     mask, sel, prob_out, sol_out, t);
  }
  k_finalize<<<1, 32, 0, stream>>>(mask, prob_out, sol_out);
}
